// Attention_58102317581028
// MI455X (gfx1250) — matrix operand order repaired and checked by reading, not compiled, not run on hardware
//
#include <hip/hip_runtime.h>
#include <math.h>

// ---------------- types ----------------
typedef __attribute__((ext_vector_type(16))) __bf16        v16bf;
typedef __attribute__((ext_vector_type(8)))  float         v8f;
typedef __attribute__((ext_vector_type(4)))  unsigned int  u32x4;

union AFragU { u32x4 q[2]; v16bf v; };

#define BB 32
#define SS 4096
#define NN 1024
#define M_TILE 64
#define N_TILE 128
#define KC 64
#define A_PITCH 72   // ushort elements per A row (64 + pad, keeps 16B align, kills bank conflicts)
#define B_PITCH 36   // u32 (K-pairs) per B column (32 + pad)

__device__ __forceinline__ unsigned short f2bf(float f) {
  unsigned int u = __float_as_uint(f);
  u += 0x7FFFu + ((u >> 16) & 1u);   // round-to-nearest-even
  return (unsigned short)(u >> 16);
}

__device__ __forceinline__ float fast_tanh(float x) {
#if __has_builtin(__builtin_amdgcn_tanhf)
  return __builtin_amdgcn_tanhf(x);   // v_tanh_f32 (gfx1250 transcendental)
#else
  return tanhf(x);
#endif
}

// ---------------- kernel 0: dec_feats[b][m] = bs[m] + sum_k dec[b,k]*Ws[m,k] ----------------
__global__ __launch_bounds__(256)
void dec_feats_kernel(const float* __restrict__ dec, const float* __restrict__ Ws,
                      const float* __restrict__ bs, float* __restrict__ out) {
  int idx = blockIdx.x * 256 + threadIdx.x;        // 32768 total
  int b = idx >> 10, m = idx & (NN - 1);
  const float4* d4 = reinterpret_cast<const float4*>(dec + (size_t)b * NN);
  const float4* w4 = reinterpret_cast<const float4*>(Ws + (size_t)m * NN);
  float acc = bs[m];
  for (int k = 0; k < NN / 4; ++k) {
    float4 a = d4[k], w = w4[k];
    acc += a.x * w.x + a.y * w.y + a.z * w.z + a.w * w.w;
  }
  out[idx] = acc;
}

// ---------------- kernel 1: fused WMMA enc_feats + tanh + v-dot -> e ----------------
// grid = (B*S)/M_TILE blocks of 128 threads (4 waves of 32; each wave owns 16 rows)
__global__ __launch_bounds__(128)
void attn_feats_wmma_kernel(const float* __restrict__ enc,
                            const float* __restrict__ Wh,
                            const float* __restrict__ bh,
                            const float* __restrict__ wc,
                            const float* __restrict__ vvec,
                            const float* __restrict__ cov,
                            const float* __restrict__ dec_feats,
                            float* __restrict__ e_out) {
  __shared__ unsigned short A_lds[M_TILE * A_PITCH];   // 9216 B
  __shared__ unsigned int   B_lds[N_TILE * B_PITCH];   // 18432 B

  const int tid  = threadIdx.x;
  const int lane = tid & 31;
  const int wid  = tid >> 5;          // wave id 0..3
  const int n16  = lane & 15;         // A-row m / B-col n / D-col n
  const int kh   = lane >> 4;         // K-half select per ISA layout

  const int row0 = blockIdx.x * M_TILE;        // 64-aligned -> single batch per WG
  const int b    = row0 >> 12;                 // row0 / 4096
  const int sb   = row0 & (SS - 1);
  const int wrow = row0 + wid * 16;            // wave's global row base

  // coverage is per-row; D element (vgpr r) maps to row wrow + r + 8*kh
  float covw[8];
#pragma unroll
  for (int r = 0; r < 8; ++r)
    covw[r] = cov[(size_t)b * SS + sb + wid * 16 + r + 8 * kh];

  float epart[8];
#pragma unroll
  for (int r = 0; r < 8; ++r) epart[r] = 0.f;

  for (int nt = 0; nt < NN / N_TILE; ++nt) {
    // per-lane column constants for this N tile (col = nt*128 + t*16 + n16)
    float base_t[8], wcv[8], vv[8];
#pragma unroll
    for (int t = 0; t < 8; ++t) {
      int col = nt * N_TILE + t * 16 + n16;
      base_t[t] = dec_feats[b * NN + col] + bh[col];
      wcv[t] = wc[col];
      vv[t]  = vvec[col];
    }

    v8f c[8];
#pragma unroll
    for (int t = 0; t < 8; ++t) c[t] = v8f{0.f,0.f,0.f,0.f,0.f,0.f,0.f,0.f};

    for (int kc = 0; kc < NN / KC; ++kc) {
      const int kb = kc * KC;
      // --- stage A: 64 rows x 64 K fp32 -> bf16, row-major pitch 72 ---
#pragma unroll
      for (int i = 0; i < 8; ++i) {
        int e4 = (tid + i * 128) * 4;
        int r_ = e4 >> 6, k_ = e4 & 63;
        const float4 f = *reinterpret_cast<const float4*>(
            enc + (size_t)(row0 + r_) * NN + kb + k_);
        unsigned short* dst = &A_lds[r_ * A_PITCH + k_];
        dst[0] = f2bf(f.x); dst[1] = f2bf(f.y);
        dst[2] = f2bf(f.z); dst[3] = f2bf(f.w);
      }
      // --- stage B: Wh[col][k..k+63] -> packed bf16 pairs, layout [col][kp] pitch 36 ---
#pragma unroll 4
      for (int i = 0; i < 32; ++i) {
        int idx  = tid + i * 128;
        int kp   = idx & 31, colL = idx >> 5;
        int colG = nt * N_TILE + colL;
        const float2 f = *reinterpret_cast<const float2*>(
            Wh + (size_t)colG * NN + kb + 2 * kp);
        B_lds[colL * B_PITCH + kp] =
            (unsigned int)f2bf(f.x) | ((unsigned int)f2bf(f.y) << 16);
      }
      // warm caches for the next K chunk (global_prefetch_b8)
      if (kc + 1 < NN / KC) {
        __builtin_prefetch(enc + (size_t)(row0 + (tid >> 1)) * NN + kb + KC + (tid & 1) * 32, 0, 3);
        __builtin_prefetch(Wh + (size_t)(nt * N_TILE + tid) * NN + kb + KC, 0, 3);
        __builtin_prefetch(Wh + (size_t)(nt * N_TILE + tid) * NN + kb + KC + 32, 0, 3);
      }
      __syncthreads();

      // --- hoist both A fragments (ks=0,1) ---
      AFragU a0, a1;
      {
        const unsigned short* ap = &A_lds[(wid * 16 + n16) * A_PITCH + kh * 8];
        a0.q[0] = *reinterpret_cast<const u32x4*>(ap);        // ks=0: K = kh*8..+7
        a0.q[1] = *reinterpret_cast<const u32x4*>(ap + 16);   //        +16
        a1.q[0] = *reinterpret_cast<const u32x4*>(ap + 32);   // ks=1
        a1.q[1] = *reinterpret_cast<const u32x4*>(ap + 48);
      }
      // --- depth-2 rotating pipeline of B fragments: load frag u+2 before wmma u ---
      AFragU bf[3];
#pragma unroll
      for (int u = 0; u < 2; ++u) {            // prologue: frags 0 and 1
        const unsigned int* bp =
            &B_lds[((u & 7) * 16 + n16) * B_PITCH + (u >> 3) * 16 + kh * 4];
        bf[u].q[0] = *reinterpret_cast<const u32x4*>(bp);
        bf[u].q[1] = *reinterpret_cast<const u32x4*>(bp + 8);
      }
#pragma unroll
      for (int u = 0; u < 16; ++u) {           // flat (ks, t) sequence
        const int ks = u >> 3, t = u & 7;
        if (u + 2 < 16) {
          const int un = u + 2, ksn = un >> 3, tn = un & 7;
          const unsigned int* bp =
              &B_lds[(tn * 16 + n16) * B_PITCH + ksn * 16 + kh * 4];
          bf[un % 3].q[0] = *reinterpret_cast<const u32x4*>(bp);
          bf[un % 3].q[1] = *reinterpret_cast<const u32x4*>(bp + 8);
        }
        c[t] = __builtin_amdgcn_wmma_f32_16x16x32_bf16(
            false, (ks ? a1.v : a0.v), false, bf[u % 3].v, (short)0, c[t],
            false, false);
      }
      __syncthreads();
    }

    // --- epilogue: feat = acc + dec_feats + bh + cov*wc ; e += v * tanh(feat) ---
#pragma unroll
    for (int t = 0; t < 8; ++t) {
#pragma unroll
      for (int r = 0; r < 8; ++r) {
        float feat = c[t][r] + base_t[t] + covw[r] * wcv[t];
        epart[r] += vv[t] * fast_tanh(feat);
      }
    }
  }

  // reduce the 16 column-lanes of each half; lane n16==0 holds row sums
#pragma unroll
  for (int r = 0; r < 8; ++r) {
    float val = epart[r];
    val += __shfl_xor(val, 8, 16);
    val += __shfl_xor(val, 4, 16);
    val += __shfl_xor(val, 2, 16);
    val += __shfl_xor(val, 1, 16);
    if (n16 == 0) e_out[wrow + kh * 8 + r] = val;
  }
}

// ---------------- kernel 2: masked renormalized softmax + coverage ----------------
__global__ __launch_bounds__(256)
void attn_softmax_kernel(const float* __restrict__ e, const int* __restrict__ lens,
                         const float* __restrict__ cov, float* __restrict__ attn,
                         float* __restrict__ cov_out) {
  const int b = blockIdx.x, tid = threadIdx.x;
  const int len = lens[b];
  __shared__ float red[256];

  float mx = -3.402823466e38f;
  for (int s = tid; s < SS; s += 256) mx = fmaxf(mx, e[(size_t)b * SS + s]);
  red[tid] = mx; __syncthreads();
  for (int off = 128; off > 0; off >>= 1) {
    if (tid < off) red[tid] = fmaxf(red[tid], red[tid + off]);
    __syncthreads();
  }
  mx = red[0]; __syncthreads();

  float sum = 0.f;
  for (int s = tid; s < SS; s += 256) {
    float p = (s < len) ? expf(e[(size_t)b * SS + s] - mx) : 0.f;
    attn[(size_t)b * SS + s] = p;
    sum += p;
  }
  red[tid] = sum; __syncthreads();
  for (int off = 128; off > 0; off >>= 1) {
    if (tid < off) red[tid] += red[tid + off];
    __syncthreads();
  }
  float inv = 1.f / red[0];
  for (int s = tid; s < SS; s += 256) {
    float a = attn[(size_t)b * SS + s] * inv;
    attn[(size_t)b * SS + s] = a;
    cov_out[(size_t)b * SS + s] = cov[(size_t)b * SS + s] + a;
  }
}

// ---------------- kernel 3: context[b][n] = sum_s attn[b,s] * enc[b,s,n] ----------------
__global__ __launch_bounds__(256)
void context_kernel(const float* __restrict__ attn, const float* __restrict__ enc,
                    float* __restrict__ ctx) {
  const int b   = blockIdx.y;
  const int col = blockIdx.x * 256 + threadIdx.x;
  __shared__ float a_sh[256];
  float acc = 0.f;
  for (int s0 = 0; s0 < SS; s0 += 256) {
    a_sh[threadIdx.x] = attn[(size_t)b * SS + s0 + threadIdx.x];
    __syncthreads();
    const float* ep = enc + ((size_t)b * SS + s0) * NN + col;
    for (int j = 0; j < 256; ++j) acc += a_sh[j] * ep[(size_t)j * NN];
    __syncthreads();
  }
  ctx[b * NN + col] = acc;
}

// ---------------- launch ----------------
extern "C" void kernel_launch(void* const* d_in, const int* in_sizes, int n_in,
                              void* d_out, int out_size, void* d_ws, size_t ws_size,
                              hipStream_t stream) {
  (void)in_sizes; (void)n_in; (void)out_size; (void)ws_size;
  const float* dec  = (const float*)d_in[0];   // [B,N]
  const float* enc  = (const float*)d_in[1];   // [B,S,N]
  const int*   lens = (const int*)  d_in[2];   // [B]
  const float* cov  = (const float*)d_in[3];   // [B,S]
  const float* Wh   = (const float*)d_in[4];   // [N,N]
  const float* bh   = (const float*)d_in[5];   // [N]
  const float* Ws   = (const float*)d_in[6];   // [N,N]
  const float* bs   = (const float*)d_in[7];   // [N]
  const float* wc   = (const float*)d_in[8];   // [N]
  const float* v    = (const float*)d_in[9];   // [N]

  float* ctx     = (float*)d_out;              // [B,N]   = 32768
  float* attn    = ctx + BB * NN;              // [B,S]   = 131072
  float* cov_out = attn + BB * SS;             // [B,S]   = 131072

  float* dec_feats = (float*)d_ws;             // 32768 floats
  float* e_buf     = dec_feats + BB * NN;      // 131072 floats

  dec_feats_kernel<<<(BB * NN) / 256, 256, 0, stream>>>(dec, Ws, bs, dec_feats);
  attn_feats_wmma_kernel<<<(BB * SS) / M_TILE, 128, 0, stream>>>(
      enc, Wh, bh, wc, v, cov, dec_feats, e_buf);
  attn_softmax_kernel<<<BB, 256, 0, stream>>>(e_buf, lens, cov, attn, cov_out);
  context_kernel<<<dim3(NN / 256, BB), 256, 0, stream>>>(attn, enc, ctx);
}
